// ThermalGuidedAttention_64415919505502
// MI455X (gfx1250) — matrix operand order repaired and checked by reading, not compiled, not run on hardware
//
#include <hip/hip_runtime.h>

// ThermalGuidedAttention for MI455X (gfx1250, wave32, WMMA + TDM).
// Pipeline:
//  k_thermal_mlp : importance[b,i] = sigmoid(w2 @ relu(w1 @ tt + b1) + b2)
//  k_interp      : thermal_mask[b,t] = linear interp of importance (TTH -> T)
//  k_proj        : Qp = f16((query@wq^T + bq) / sqrt(HD)), Kp = f16(key@wk^T + bk)   [WMMA]
//  k_vt          : Vt[b,d,t] = f16(value[b,t,d])
//  k_attn        : per (b, 16-row q tile): TDM-stage Q panel to LDS, per-head WMMA scores
//                  -> softmax -> head mean (in LDS) -> re-softmax (+thermal) -> guided (NT store)
//                  -> guided@V via WMMA -> guided_out

constexpr int B = 2, T = 2048, TTH = 512, D = 1024, H = 16, HD = 64;

typedef __attribute__((ext_vector_type(16))) _Float16 v16h;
typedef __attribute__((ext_vector_type(8)))  _Float16 v8h;
typedef __attribute__((ext_vector_type(8)))  float    v8f;
typedef __attribute__((ext_vector_type(4)))  unsigned int u32x4;
typedef __attribute__((ext_vector_type(8)))  int      i32x8;
typedef __attribute__((ext_vector_type(4)))  int      i32x4;

#define DEV_INLINE __device__ __forceinline__

DEV_INLINE v16h combine8(v8h lo, v8h hi) {
  return __builtin_shufflevector(lo, hi, 0,1,2,3,4,5,6,7,8,9,10,11,12,13,14,15);
}

DEV_INLINE v8f wmma_f16(v16h a, v16h b, v8f c) {
  // D = A(16x32) * B(32x16) + C, f32 accum
  return __builtin_amdgcn_wmma_f32_16x16x32_f16(false, a, false, b, (short)0, c, false, false);
}

// A-fragment (16x32 MxK, f16) from f32 row-major src; base pre-offset to tile row 0.
// lane<16: M=lane, K={0..7,16..23}; lane>=16: M=lane-16, K={8..15,24..31}
DEV_INLINE v16h a_frag_f32(const float* base, int ld, int k0, int lane) {
  int r  = lane & 15;
  int kb = (lane >> 4) << 3;
  const float* p = base + (size_t)r * ld + k0 + kb;
  v16h a;
#pragma unroll
  for (int i = 0; i < 8; ++i) a[i] = (_Float16)p[i];
#pragma unroll
  for (int i = 0; i < 8; ++i) a[8 + i] = (_Float16)p[16 + i];
  return a;
}

// B-fragment (32x16 KxN, f16) with B[k][n] = W[n][k], W f32 row-major.
DEV_INLINE v16h b_frag_f32T(const float* W, int ld, int n0, int k0, int lane) {
  int n  = n0 + (lane & 15);
  int kb = (lane >> 4) << 3;
  const float* p = W + (size_t)n * ld + k0 + kb;
  v16h b;
#pragma unroll
  for (int i = 0; i < 8; ++i) b[i] = (_Float16)p[i];
#pragma unroll
  for (int i = 0; i < 8; ++i) b[8 + i] = (_Float16)p[16 + i];
  return b;
}

// f16-source variants (works for global or LDS pointers; 16B/32B aligned)
DEV_INLINE v16h a_frag_h(const _Float16* base, int ld, int k0, int lane) {
  int r  = lane & 15;
  int kb = (lane >> 4) << 3;
  const _Float16* p = base + (size_t)r * ld + k0 + kb;
  return combine8(*(const v8h*)p, *(const v8h*)(p + 16));
}
DEV_INLINE v16h b_frag_hT(const _Float16* Wt, int ld, int n0, int k0, int lane) {
  int n  = n0 + (lane & 15);
  int kb = (lane >> 4) << 3;
  const _Float16* p = Wt + (size_t)n * ld + k0 + kb;
  return combine8(*(const v8h*)p, *(const v8h*)(p + 16));
}

// TDM: DMA a [rows x 1024] f16 row-major tile (row stride 1024 elems) into LDS.
// D# per CDNA5 ISA ch.8: group0 = {flags, lds_addr, global_addr, type=2},
// group1 = {mask/data_size, tensor dims, tile dims, dim0 stride}.
DEV_INLINE void tdm_load_tile_f16(const _Float16* gsrc, unsigned lds_off, int rows) {
  unsigned long long ga = (unsigned long long)(uintptr_t)gsrc;
  u32x4 g0 = {};
  g0[0] = 1u;                                  // count=1, user mode, no gather
  g0[1] = lds_off;                             // lds_addr (bytes)
  g0[2] = (unsigned)ga;                        // global_addr[31:0]
  g0[3] = (unsigned)(ga >> 32) | (2u << 30);   // global_addr[56:32] | type=2
  i32x8 g1 = {};
  g1[0] = (1 << 16);                           // workgroup_mask=0, data_size=1 (2B)
  g1[1] = (int)((1024u & 0xFFFFu) << 16);      // tensor_dim0 = 1024 (lo16 in w1[31:16])
  g1[2] = (int)(((unsigned)rows & 0xFFFFu) << 16); // tensor_dim0 hi=0 | tensor_dim1 = rows
  g1[3] = (int)((1024u & 0xFFFFu) << 16);      // tensor_dim1 hi=0 | tile_dim0 = 1024
  g1[4] = rows;                                // tile_dim1 = rows, tile_dim2 = 0
  g1[5] = 1024;                                // tensor_dim0_stride[31:0] = 1024 elems
  g1[6] = 0;                                   // stride0 hi | stride1 lo
  g1[7] = 0;
  i32x4 gz = {};
#if __clang_major__ >= 23
  i32x8 gz8 = {};
  __builtin_amdgcn_tensor_load_to_lds(g0, g1, gz, gz, gz8, 0);
#else
  __builtin_amdgcn_tensor_load_to_lds(g0, g1, gz, gz, 0);
#endif
}

// ---------------------------------------------------------------- thermal MLP
__global__ __launch_bounds__(256) void k_thermal_mlp(
    const float* __restrict__ tt, const float* __restrict__ w1,
    const float* __restrict__ b1, const float* __restrict__ w2,
    const float* __restrict__ b2, float* __restrict__ imp) {
  __shared__ float red[256];
  int tok = blockIdx.x;                      // b*TTH + i
  const float* x = tt + (size_t)tok * D;
  int tid = threadIdx.x;
  float partial = 0.f;
#pragma unroll
  for (int jj = 0; jj < 2; ++jj) {
    int j = tid + jj * 256;                  // hidden unit (D/2 = 512)
    const float* wr = w1 + (size_t)j * D;
    float acc = b1[j];
    for (int d = 0; d < D; d += 4)
      acc += x[d]*wr[d] + x[d+1]*wr[d+1] + x[d+2]*wr[d+2] + x[d+3]*wr[d+3];
    float hr = acc > 0.f ? acc : 0.f;
    partial += hr * w2[j];
  }
  red[tid] = partial;
  __syncthreads();
  for (int off = 128; off > 0; off >>= 1) {
    if (tid < off) red[tid] += red[tid + off];
    __syncthreads();
  }
  if (tid == 0) {
    float s = red[0] + b2[0];
    imp[tok] = 1.f / (1.f + __expf(-s));
  }
}

// ------------------------------------------------------- linear interpolation
__global__ __launch_bounds__(256) void k_interp(const float* __restrict__ imp,
                                                float* __restrict__ tm) {
  int idx = blockIdx.x * 256 + threadIdx.x;  // b*T + t
  int b = idx / T, t = idx % T;
  float scale = (float)TTH / (float)T;
  float c = ((float)t + 0.5f) * scale - 0.5f;
  c = fminf(fmaxf(c, 0.f), (float)(TTH - 1));
  int x0 = (int)floorf(c);
  int x1 = min(x0 + 1, TTH - 1);
  float w = c - (float)x0;
  tm[idx] = imp[b * TTH + x0] * (1.f - w) + imp[b * TTH + x1] * w;
}

// ------------------------------------------------------ Q/K projection (WMMA)
__global__ __launch_bounds__(32) void k_proj(
    const float* __restrict__ query, const float* __restrict__ key,
    const float* __restrict__ in_w, const float* __restrict__ in_b,
    _Float16* __restrict__ Qp, _Float16* __restrict__ Kp) {
  int lane  = threadIdx.x;
  int n0    = blockIdx.x * 16;
  int m0    = blockIdx.y * 16;
  int which = blockIdx.z / B;                // 0 = Q, 1 = K
  int b     = blockIdx.z % B;
  const float* x    = (which ? key : query) + (size_t)(b * T + m0) * D;
  const float* W    = in_w + (size_t)which * D * D;
  const float* bias = in_b + which * D;
  _Float16*    out  = which ? Kp : Qp;
  float scale = which ? 1.f : 0.125f;        // 1/sqrt(HD)=1/8 folded into Q
  v8f acc = {};
  for (int k0 = 0; k0 < D; k0 += 32) {
    v16h a  = a_frag_f32(x, D, k0, lane);
    v16h bm = b_frag_f32T(W, D, n0, k0, lane);
    acc = wmma_f16(a, bm, acc);
  }
  int n = n0 + (lane & 15);
  float bn = bias[n];
  int mbase = m0 + ((lane >> 4) << 3);
#pragma unroll
  for (int r = 0; r < 8; ++r) {
    float v = (acc[r] + bn) * scale;
    out[(size_t)(b * T + mbase + r) * D + n] = (_Float16)v;
  }
}

// --------------------------------------------- V transpose + f16 convert (Vt)
__global__ __launch_bounds__(256) void k_vt(const float* __restrict__ value,
                                            _Float16* __restrict__ Vt) {
  __shared__ float tile[16][17];
  int b  = blockIdx.z;
  int t0 = blockIdx.x * 16;
  int d0 = blockIdx.y * 16;
  int tx = threadIdx.x, ty = threadIdx.y;
  tile[ty][tx] = value[(size_t)(b * T + t0 + ty) * D + d0 + tx];
  __syncthreads();
  Vt[(size_t)(b * D + d0 + ty) * T + t0 + tx] = (_Float16)tile[tx][ty];
}

// ------------------------------------------------- fused attention (per tile)
__global__ __launch_bounds__(256) void k_attn(
    const _Float16* __restrict__ Qp, const _Float16* __restrict__ Kp,
    const _Float16* __restrict__ Vt, const float* __restrict__ tm,
    float* __restrict__ outO, float* __restrict__ outG) {
  extern __shared__ char smem[];
  _Float16* qt = (_Float16*)smem;              // [16][D] f16 Q panel   (32 KB)
  float* sc  = (float*)(smem + 32 * 1024);     // [16][T] per-head scores (128 KB)
  float* avg = sc + 16 * T;                    // [16][T] head-avg / guided (128 KB)
  float* red = avg + 16 * T;                   // [256] reduction scratch

  int tid  = threadIdx.x;
  int wave = tid >> 5, lane = tid & 31;
  int qb = blockIdx.x % (T / 16);
  int b  = blockIdx.x / (T / 16);
  int q0 = qb * 16;

  // TDM: stage Q panel [16 x D] f16 into LDS (one DMA per block, wave 0 issues)
  if (wave == 0) {
    tdm_load_tile_f16(Qp + (size_t)(b * T + q0) * D,
                      (unsigned)(uintptr_t)qt, 16);
    __builtin_amdgcn_s_wait_tensorcnt(0);
  }
  for (int i = tid; i < 16 * T; i += 256) avg[i] = 0.f;
  __syncthreads();

  int row = tid >> 4;           // 0..15 (softmax row per 16-thread group)
  int sub = tid & 15;
  int cN  = lane & 15;          // C/D fragment column
  int cR  = (lane >> 4) << 3;   // C/D fragment row base

  for (int h = 0; h < H; ++h) {
    // scores[16][T] for head h : wave w owns column tiles w, w+8, ...
    v16h a0 = a_frag_h(qt, D, h * HD, lane);       // from LDS (ds_load)
    v16h a1 = a_frag_h(qt, D, h * HD + 32, lane);
    for (int ct = wave; ct < T / 16; ct += 8) {
      int n0 = ct * 16;
      const _Float16* Krow = Kp + (size_t)(b * T + n0) * D + h * HD;
      int pn = n0 + 128 < T ? n0 + 128 : n0;
      __builtin_prefetch(Kp + (size_t)(b * T + pn) * D + h * HD, 0, 1);
      v16h b0 = b_frag_hT(Krow, D, 0, 0, lane);
      v16h b1 = b_frag_hT(Krow, D, 0, 32, lane);
      v8f acc = {};
      acc = wmma_f16(a0, b0, acc);
      acc = wmma_f16(a1, b1, acc);
#pragma unroll
      for (int r = 0; r < 8; ++r)
        sc[(cR + r) * T + n0 + cN] = acc[r];
    }
    __syncthreads();

    // row softmax, accumulate into avg (each thread: fixed row, cols = sub mod 16)
    float mx = -3.4e38f;
    for (int c = sub; c < T; c += 16) mx = fmaxf(mx, sc[row * T + c]);
    red[tid] = mx;
    for (int off = 8; off > 0; off >>= 1) {
      __syncthreads();
      if (sub < off) red[tid] = fmaxf(red[tid], red[tid + off]);
    }
    __syncthreads();
    mx = red[row * 16];
    float sum = 0.f;
    for (int c = sub; c < T; c += 16) {
      float e = __expf(sc[row * T + c] - mx);
      sc[row * T + c] = e;
      sum += e;
    }
    red[tid] = sum;
    for (int off = 8; off > 0; off >>= 1) {
      __syncthreads();
      if (sub < off) red[tid] += red[tid + off];
    }
    __syncthreads();
    float inv = 1.f / red[row * 16];
    for (int c = sub; c < T; c += 16) avg[row * T + c] += sc[row * T + c] * inv;
    __syncthreads();
  }

  // guided = softmax(avg/H + 0.1*thermal_mask[q]) ; emit guided to outG (NT stores)
  {
    float tbias = 0.1f * tm[b * T + q0 + row];
    float mx = -3.4e38f;
    for (int c = sub; c < T; c += 16) {
      float v = avg[row * T + c] * (1.f / H) + tbias;
      avg[row * T + c] = v;
      mx = fmaxf(mx, v);
    }
    red[tid] = mx;
    for (int off = 8; off > 0; off >>= 1) {
      __syncthreads();
      if (sub < off) red[tid] = fmaxf(red[tid], red[tid + off]);
    }
    __syncthreads();
    mx = red[row * 16];
    float sum = 0.f;
    for (int c = sub; c < T; c += 16) {
      float e = __expf(avg[row * T + c] - mx);
      avg[row * T + c] = e;
      sum += e;
    }
    red[tid] = sum;
    for (int off = 8; off > 0; off >>= 1) {
      __syncthreads();
      if (sub < off) red[tid] += red[tid + off];
    }
    __syncthreads();
    float inv = 1.f / red[row * 16];
    float* Grow = outG + (size_t)(b * T + q0 + row) * T;
    for (int c = sub; c < T; c += 16) {
      float g = avg[row * T + c] * inv;
      avg[row * T + c] = g;
      __builtin_nontemporal_store(g, Grow + c);   // streamed 33MB, written once
    }
    __syncthreads();
  }

  // guided_out[16, D] = guided[16, T] @ value  (B-frags from Vt, contiguous)
  {
    v8f zero = {};
    v8f acc[8];
#pragma unroll
    for (int i = 0; i < 8; ++i) acc[i] = zero;
    int r  = lane & 15;
    int kb = (lane >> 4) << 3;
    const _Float16* Vb = Vt + (size_t)b * D * T;
    for (int k0 = 0; k0 < T; k0 += 32) {
      const float* ap = avg + (size_t)r * T + k0 + kb;
      v16h a;
#pragma unroll
      for (int i = 0; i < 8; ++i) a[i] = (_Float16)ap[i];
#pragma unroll
      for (int i = 0; i < 8; ++i) a[8 + i] = (_Float16)ap[16 + i];
#pragma unroll
      for (int ti = 0; ti < 8; ++ti) {
        int n0 = (wave + ti * 8) * 16;
        v16h bm = b_frag_hT(Vb, T, n0, k0, lane);
        acc[ti] = wmma_f16(a, bm, acc[ti]);
      }
    }
#pragma unroll
    for (int ti = 0; ti < 8; ++ti) {
      int n0 = (wave + ti * 8) * 16;
#pragma unroll
      for (int rr = 0; rr < 8; ++rr)
        outO[(size_t)(b * T + q0 + cR + rr) * D + n0 + cN] = acc[ti][rr];
    }
  }
}

// ----------------------------------------------------------------------------
extern "C" void kernel_launch(void* const* d_in, const int* in_sizes, int n_in,
                              void* d_out, int out_size, void* d_ws, size_t ws_size,
                              hipStream_t stream) {
  const float* query = (const float*)d_in[0];
  const float* key   = (const float*)d_in[1];
  const float* value = (const float*)d_in[2];
  const float* tt    = (const float*)d_in[3];
  const float* in_w  = (const float*)d_in[4];
  const float* in_b  = (const float*)d_in[5];
  const float* w1    = (const float*)d_in[6];
  const float* b1    = (const float*)d_in[7];
  const float* w2    = (const float*)d_in[8];
  const float* b2    = (const float*)d_in[9];

  float* outO = (float*)d_out;                        // [B,T,D]
  float* outG = (float*)d_out + (size_t)B * T * D;    // [B,T,T]

  // workspace: Qp/Kp/Vt f16 (8MB each) + importance + thermal mask
  _Float16* Qp = (_Float16*)d_ws;
  _Float16* Kp = Qp + (size_t)B * T * D;
  _Float16* Vt = Kp + (size_t)B * T * D;
  float* imp   = (float*)(Vt + (size_t)B * T * D);
  float* tmask = imp + B * TTH;

  k_thermal_mlp<<<B * TTH, 256, 0, stream>>>(tt, w1, b1, w2, b2, imp);
  k_interp<<<(B * T) / 256, 256, 0, stream>>>(imp, tmask);
  k_proj<<<dim3(D / 16, T / 16, 2 * B), 32, 0, stream>>>(query, key, in_w, in_b, Qp, Kp);
  k_vt<<<dim3(T / 16, D / 16, B), dim3(16, 16), 0, stream>>>(value, Vt);

  // LDS: 32KB Q panel + 128KB scores + 128KB avg + 1KB scratch = 289KB < 320KB/WGP
  size_t smem = 32 * 1024 + (size_t)(2 * 16 * T + 256) * sizeof(float);
  k_attn<<<B * (T / 16), 256, smem, stream>>>(Qp, Kp, Vt, tmask, outO, outG);
}
